// Attention_50775103373799
// MI455X (gfx1250) — matrix operand order repaired and checked by reading, not compiled, not run on hardware
//
#include <hip/hip_runtime.h>
#include <hip/hip_bf16.h>
#include <cstdint>

typedef __bf16 bf16;
typedef __attribute__((ext_vector_type(16))) __bf16 v16bf;
typedef __attribute__((ext_vector_type(4)))  __bf16 v4bf;
typedef __attribute__((ext_vector_type(8)))  float  v8f;

constexpr int B_  = 4;
constexpr int S_  = 2048;
constexpr int DIM_ = 2048;
constexpr int NH_ = 16;
constexpr int HD_ = 128;
constexpr int N3_ = 3 * DIM_;   // 6144
constexpr int LDK = 40;         // padded LDS row stride (halfs) for 32-wide K tiles

#define WMMA_BF16(a, b, c) \
  __builtin_amdgcn_wmma_f32_16x16x32_bf16(false, (a), false, (b), (short)0, (c), false, false)

// ---------------- async memory->LDS copy (CDNA5), guarded -------------------
#if defined(__has_builtin)
# if __has_builtin(__builtin_amdgcn_global_load_async_to_lds_b128)
#  define ASYNC_COPY_OK 1
# endif
#endif

typedef int v4i __attribute__((ext_vector_type(4)));
typedef v4i as1_v4i __attribute__((address_space(1)));   // global
typedef v4i as3_v4i __attribute__((address_space(3)));   // LDS

__device__ __forceinline__ void async_cp_b128(const void* g, void* l) {
#ifdef ASYNC_COPY_OK
  // global addr == flat addr numerically; LDS offset == low 32 bits of flat LDS addr
  __builtin_amdgcn_global_load_async_to_lds_b128(
      (as1_v4i*)(unsigned long long)(uintptr_t)g,
      (as3_v4i*)(unsigned int)(uintptr_t)l,
      0, 0);
#else
  *(uint4*)l = *(const uint4*)g;
#endif
}
__device__ __forceinline__ void async_fence() {
#ifdef ASYNC_COPY_OK
# if __has_builtin(__builtin_amdgcn_s_wait_asynccnt)
  __builtin_amdgcn_s_wait_asynccnt(0);
# else
  asm volatile("s_wait_asynccnt 0" ::: "memory");
# endif
#endif
}

// ---- fragment loaders (ISA 7.12.2 layouts), p = start of this lane's row ----
// A matrix 16x32 bf16: lane m = lane&15; lanes<16 hold K {0..7,16..23}, lanes>=16 hold K {8..15,24..31}
__device__ __forceinline__ v16bf frag_a_row(const bf16* p, int lane) {
  const int kb = (lane & 16) ? 8 : 0;
  union { v16bf v; uint4 u[2]; } r;
  r.u[0] = *(const uint4*)(p + kb);
  r.u[1] = *(const uint4*)(p + kb + 16);
  return r.v;
}
// B matrix 32x16 bf16 stored as [n][k] rows: lane n = lane&15; lanes<16 hold K {0..7,16..23}, lanes>=16 hold K {8..15,24..31}
__device__ __forceinline__ v16bf frag_b_row(const bf16* p, int lane) {
  const int kb = (lane & 16) ? 8 : 0;
  union { v16bf v; uint4 u[2]; } r;
  r.u[0] = *(const uint4*)(p + kb);
  r.u[1] = *(const uint4*)(p + kb + 16);
  return r.v;
}

__device__ __forceinline__ v8f vzero8() {
  v8f z;
#pragma unroll
  for (int i = 0; i < 8; ++i) z[i] = 0.f;
  return z;
}

// ============================================================================
// Kernel 1: xqkv = x @ Wqkv + bqkv
//   Q,K -> [B,NH,S,HD] bf16 ; V -> [B,NH,HD,S] bf16 (pre-transposed for P@V)
// ============================================================================
__global__ __launch_bounds__(256)
void qkv_gemm_kernel(const float* __restrict__ x, const float* __restrict__ W,
                     const float* __restrict__ bias,
                     bf16* __restrict__ qws, bf16* __restrict__ kws, bf16* __restrict__ vtws) {
  __shared__ __align__(16) bf16 ldsA[128][LDK];   // [m][k]
  __shared__ __align__(16) bf16 ldsB[128][LDK];   // [n][k] (transposed W tile)
  const int tid = threadIdx.x, lane = tid & 31, wave = tid >> 5;
  const int rb = blockIdx.y * 128;
  const int cb = blockIdx.x * 128;
  const int mw = wave >> 2;   // 0..1 -> 64 rows each
  const int nw = wave & 3;    // 0..3 -> 32 cols each

  v8f acc[4][2];
#pragma unroll
  for (int mt = 0; mt < 4; ++mt)
#pragma unroll
    for (int nt = 0; nt < 2; ++nt) acc[mt][nt] = vzero8();

  for (int kt = 0; kt < DIM_ / 32; ++kt) {
    const int k0 = kt * 32;
    // A tile: 128x32 f32 -> bf16 LDS
    {
      const int f = tid & 7;      // float4 slot in row
      const int r0 = tid >> 3;    // 0..31
#pragma unroll
      for (int i = 0; i < 4; ++i) {
        const int r = r0 + 32 * i;
        const float4 d = *(const float4*)(x + (size_t)(rb + r) * DIM_ + k0 + f * 4);
        v4bf h;
        h[0] = (bf16)d.x; h[1] = (bf16)d.y; h[2] = (bf16)d.z; h[3] = (bf16)d.w;
        *(v4bf*)&ldsA[r][f * 4] = h;
      }
    }
    // W tile: 32x128 f32 -> bf16 LDS transposed [n][k]
    {
      const int n4 = tid & 31;    // float4 col
      const int kk0 = tid >> 5;   // 0..7
#pragma unroll
      for (int i = 0; i < 4; ++i) {
        const int kk = kk0 + 8 * i;
        const float4 d = *(const float4*)(W + (size_t)(k0 + kk) * N3_ + cb + n4 * 4);
        ldsB[n4 * 4 + 0][kk] = (bf16)d.x;
        ldsB[n4 * 4 + 1][kk] = (bf16)d.y;
        ldsB[n4 * 4 + 2][kk] = (bf16)d.z;
        ldsB[n4 * 4 + 3][kk] = (bf16)d.w;
      }
    }
    if (kt + 1 < DIM_ / 32) {  // WGP-scope prefetch of next tiles
      __builtin_prefetch(x + (size_t)(rb + (tid >> 3)) * DIM_ + k0 + 32, 0, 3);
      __builtin_prefetch(W + (size_t)(k0 + 32 + (tid >> 5)) * N3_ + cb, 0, 3);
    }
    __syncthreads();

    v16bf afr[4], bfr[2];
#pragma unroll
    for (int mt = 0; mt < 4; ++mt)
      afr[mt] = frag_a_row(&ldsA[mw * 64 + mt * 16 + (lane & 15)][0], lane);
#pragma unroll
    for (int nt = 0; nt < 2; ++nt)
      bfr[nt] = frag_b_row(&ldsB[nw * 32 + nt * 16 + (lane & 15)][0], lane);
#pragma unroll
    for (int mt = 0; mt < 4; ++mt)
#pragma unroll
      for (int nt = 0; nt < 2; ++nt)
        acc[mt][nt] = WMMA_BF16(afr[mt], bfr[nt], acc[mt][nt]);
    __syncthreads();
  }

  // epilogue: bias + scatter (V goes out transposed, packed b128 stores)
#pragma unroll
  for (int nt = 0; nt < 2; ++nt) {
    const int col = cb + nw * 32 + nt * 16 + (lane & 15);
    const float bv = bias[col];
    const int which = col >> 11;         // 0=Q 1=K 2=V
    const int d = col & (DIM_ - 1);
    const int h = d >> 7, hd = d & (HD_ - 1);
    if (which == 2) {
      // V^T: [B,NH,HD,S]; the lane's 8 rows are 8 consecutive s -> one b128 store
#pragma unroll
      for (int mt = 0; mt < 4; ++mt) {
        const int row = rb + mw * 64 + mt * 16 + ((lane & 16) ? 8 : 0);
        const int b = row >> 11, s0 = row & (S_ - 1);
        union { uint4 u; bf16 hh[8]; } pk;
#pragma unroll
        for (int r = 0; r < 8; ++r) pk.hh[r] = (bf16)(acc[mt][nt][r] + bv);
        *(uint4*)(vtws + (((size_t)(b * NH_ + h)) * HD_ + hd) * S_ + s0) = pk.u;
      }
    } else {
      bf16* base = (which == 0) ? qws : kws;
#pragma unroll
      for (int mt = 0; mt < 4; ++mt) {
#pragma unroll
        for (int r = 0; r < 8; ++r) {
          const int row = rb + mw * 64 + mt * 16 + r + ((lane & 16) ? 8 : 0);
          const int b = row >> 11, s = row & (S_ - 1);
          base[(((size_t)(b * NH_ + h)) * S_ + s) * HD_ + hd] = (bf16)(acc[mt][nt][r] + bv);
        }
      }
    }
  }
}

// ============================================================================
// Kernel 2: flash attention per (b,h); 128 query rows per block, 8 waves x 16 rows
// ============================================================================
__global__ __launch_bounds__(256)
void attn_kernel(const bf16* __restrict__ qws, const bf16* __restrict__ kws,
                 const bf16* __restrict__ vtws, bf16* __restrict__ ows) {
  __shared__ __align__(16) bf16 ldsK[64][HD_ + 8];     // [key][hd]
  __shared__ __align__(16) bf16 ldsV[HD_][64 + 8];     // [hd][key] (V^T rows, verbatim)
  __shared__ __align__(16) bf16 ldsP[8][16][64 + 8];   // per-wave P tile
  const int tid = threadIdx.x, lane = tid & 31, wave = tid >> 5;
  const int bh = blockIdx.y;
  const int qb = blockIdx.x * 128;
  const int row0 = qb + wave * 16;
  const float scale = 0.08838834764831845f;            // 1/sqrt(128)
  const size_t base = (size_t)bh * S_ * HD_;

  // persistent Q fragments: 16 rows x 128 (4 K-chunks of 32)
  v16bf qf[4];
  {
    const bf16* qrow = qws + base + (size_t)(row0 + (lane & 15)) * HD_;
#pragma unroll
    for (int kc = 0; kc < 4; ++kc) qf[kc] = frag_a_row(qrow + kc * 32, lane);
  }

  v8f accO[8];
#pragma unroll
  for (int i = 0; i < 8; ++i) accO[i] = vzero8();
  float mrow[8], lrow[8];
#pragma unroll
  for (int r = 0; r < 8; ++r) { mrow[r] = -3.0e38f; lrow[r] = 0.f; }

  const int jend = qb + 128;
  for (int j0 = 0; j0 < jend; j0 += 64) {
    // cooperative K/V tile copies into LDS (async-to-LDS when available)
    {
      const bf16* kp = kws + base + (size_t)j0 * HD_;
#pragma unroll
      for (int i = 0; i < 4; ++i) {
        const int idx = tid + 256 * i;              // 0..1023 uint4s
        const int key = idx >> 4, c = idx & 15;     // 16 uint4 per 128-half row
        async_cp_b128(kp + (size_t)key * HD_ + c * 8, &ldsK[key][c * 8]);
      }
      const bf16* vp = vtws + base + j0;            // V^T rows: [hd][s]
#pragma unroll
      for (int i = 0; i < 2; ++i) {
        const int idx = tid + 256 * i;              // 0..511 pairs of uint4
        const int hd = idx >> 2, c = idx & 3;       // 4 x 16-half chunks per row
        async_cp_b128(vp + (size_t)hd * S_ + c * 16, &ldsV[hd][c * 16]);
        async_cp_b128(vp + (size_t)hd * S_ + c * 16 + 8, &ldsV[hd][c * 16 + 8]);
      }
    }
    async_fence();
    __syncthreads();

    if (j0 <= row0 + 15) {
      // S = Q @ K^T  (4 n-subtiles of 16 keys)
      v8f sc[4];
#pragma unroll
      for (int nt = 0; nt < 4; ++nt) {
        v8f s = vzero8();
        const bf16* krow = &ldsK[nt * 16 + (lane & 15)][0];
#pragma unroll
        for (int kc = 0; kc < 4; ++kc) {
          v16bf bk = frag_b_row(krow + kc * 32, lane);
          s = WMMA_BF16(qf[kc], bk, s);
        }
        sc[nt] = s;
      }
      // scale + causal mask + online softmax (rows live in 16-lane halves)
      float mnew[8];
#pragma unroll
      for (int r = 0; r < 8; ++r) {
        const int row = row0 + r + ((lane & 16) ? 8 : 0);
        float mx = -3.0e38f;
#pragma unroll
        for (int nt = 0; nt < 4; ++nt) {
          const int col = j0 + nt * 16 + (lane & 15);
          float v = sc[nt][r] * scale + ((col <= row) ? 0.f : -3.0e38f);
          sc[nt][r] = v;
          mx = fmaxf(mx, v);
        }
#pragma unroll
        for (int off = 1; off < 16; off <<= 1)
          mx = fmaxf(mx, __shfl_xor(mx, off, 32));
        mnew[r] = fmaxf(mrow[r], mx);
      }
#pragma unroll
      for (int r = 0; r < 8; ++r) {
        const float alpha = __expf(mrow[r] - mnew[r]);
        float rs = 0.f;
#pragma unroll
        for (int nt = 0; nt < 4; ++nt) {
          float p = __expf(sc[nt][r] - mnew[r]);
          sc[nt][r] = p;
          rs += p;
        }
#pragma unroll
        for (int off = 1; off < 16; off <<= 1)
          rs += __shfl_xor(rs, off, 32);
        lrow[r] = lrow[r] * alpha + rs;
        mrow[r] = mnew[r];
#pragma unroll
        for (int ht = 0; ht < 8; ++ht) accO[ht][r] *= alpha;
      }
      // C-layout P -> bf16 LDS (per-wave region, in-wave DS ordering)
#pragma unroll
      for (int nt = 0; nt < 4; ++nt) {
        const int col = nt * 16 + (lane & 15);
#pragma unroll
        for (int r = 0; r < 8; ++r) {
          const int rl = r + ((lane & 16) ? 8 : 0);
          ldsP[wave][rl][col] = (bf16)sc[nt][r];
        }
      }
      // O += P @ V
#pragma unroll
      for (int ks = 0; ks < 2; ++ks) {
        v16bf pf = frag_a_row(&ldsP[wave][lane & 15][ks * 32], lane);
#pragma unroll
        for (int ht = 0; ht < 8; ++ht) {
          v16bf bv = frag_b_row(&ldsV[ht * 16 + (lane & 15)][ks * 32], lane);
          accO[ht] = WMMA_BF16(pf, bv, accO[ht]);
        }
      }
    }
    __syncthreads();
  }

  // normalize + store O in [B,NH,S,HD] bf16
#pragma unroll
  for (int r = 0; r < 8; ++r) {
    const float inv = (lrow[r] > 0.f) ? 1.f / lrow[r] : 0.f;
    const int row = row0 + r + ((lane & 16) ? 8 : 0);
    bf16* orow = ows + base + (size_t)row * HD_;
#pragma unroll
    for (int ht = 0; ht < 8; ++ht)
      orow[ht * 16 + (lane & 15)] = (bf16)(accO[ht][r] * inv);
  }
}

// ============================================================================
// Kernel 3: out = O @ Wproj + bproj   (O gathered from [B,NH,S,HD] bf16)
// ============================================================================
__global__ __launch_bounds__(256)
void proj_gemm_kernel(const bf16* __restrict__ ows, const float* __restrict__ W,
                      const float* __restrict__ bias, float* __restrict__ out) {
  __shared__ __align__(16) bf16 ldsA[128][LDK];
  __shared__ __align__(16) bf16 ldsB[128][LDK];
  const int tid = threadIdx.x, lane = tid & 31, wave = tid >> 5;
  const int rb = blockIdx.y * 128;
  const int cb = blockIdx.x * 128;
  const int mw = wave >> 2, nw = wave & 3;

  v8f acc[4][2];
#pragma unroll
  for (int mt = 0; mt < 4; ++mt)
#pragma unroll
    for (int nt = 0; nt < 2; ++nt) acc[mt][nt] = vzero8();

  for (int kt = 0; kt < DIM_ / 32; ++kt) {
    const int k0 = kt * 32;
    const int h = k0 >> 7, hd0 = k0 & (HD_ - 1);   // 32-chunk stays inside one head
    // A tile: bf16 gather from [B,NH,S,HD] (async-to-LDS when available)
    {
#pragma unroll
      for (int i = 0; i < 2; ++i) {
        const int idx = tid + 256 * i;             // 0..511 uint4s
        const int r = idx >> 2, c8 = idx & 3;
        const int row = rb + r;
        const int b = row >> 11, s = row & (S_ - 1);
        const bf16* p = ows + (((size_t)(b * NH_ + h)) * S_ + s) * HD_ + hd0 + c8 * 8;
        async_cp_b128(p, &ldsA[r][c8 * 8]);
      }
    }
    // Wproj tile: 32x128 f32 -> bf16 transposed [n][k]
    {
      const int n4 = tid & 31, kk0 = tid >> 5;
#pragma unroll
      for (int i = 0; i < 4; ++i) {
        const int kk = kk0 + 8 * i;
        const float4 d = *(const float4*)(W + (size_t)(k0 + kk) * DIM_ + cb + n4 * 4);
        ldsB[n4 * 4 + 0][kk] = (bf16)d.x;
        ldsB[n4 * 4 + 1][kk] = (bf16)d.y;
        ldsB[n4 * 4 + 2][kk] = (bf16)d.z;
        ldsB[n4 * 4 + 3][kk] = (bf16)d.w;
      }
    }
    if (kt + 1 < DIM_ / 32)
      __builtin_prefetch(W + (size_t)(k0 + 32 + (tid >> 5)) * DIM_ + cb, 0, 3);
    async_fence();
    __syncthreads();

    v16bf afr[4], bfr[2];
#pragma unroll
    for (int mt = 0; mt < 4; ++mt)
      afr[mt] = frag_a_row(&ldsA[mw * 64 + mt * 16 + (lane & 15)][0], lane);
#pragma unroll
    for (int nt = 0; nt < 2; ++nt)
      bfr[nt] = frag_b_row(&ldsB[nw * 32 + nt * 16 + (lane & 15)][0], lane);
#pragma unroll
    for (int mt = 0; mt < 4; ++mt)
#pragma unroll
      for (int nt = 0; nt < 2; ++nt)
        acc[mt][nt] = WMMA_BF16(afr[mt], bfr[nt], acc[mt][nt]);
    __syncthreads();
  }

#pragma unroll
  for (int nt = 0; nt < 2; ++nt) {
    const int col = cb + nw * 32 + nt * 16 + (lane & 15);
    const float bv = bias[col];
#pragma unroll
    for (int mt = 0; mt < 4; ++mt) {
#pragma unroll
      for (int r = 0; r < 8; ++r) {
        const int row = rb + mw * 64 + mt * 16 + r + ((lane & 16) ? 8 : 0);
        out[(size_t)row * DIM_ + col] = acc[mt][nt][r] + bv;   // coalesced f32 stores
      }
    }
  }
}

// ============================================================================
extern "C" void kernel_launch(void* const* d_in, const int* in_sizes, int n_in,
                              void* d_out, int out_size, void* d_ws, size_t ws_size,
                              hipStream_t stream) {
  (void)in_sizes; (void)n_in; (void)out_size; (void)ws_size;
  const float* x     = (const float*)d_in[0];
  // d_in[1] = additive mask (causal; recomputed inline)
  const float* Wqkv  = (const float*)d_in[2];
  const float* bqkv  = (const float*)d_in[3];
  const float* Wproj = (const float*)d_in[4];
  const float* bproj = (const float*)d_in[5];
  // d_in[6] = start_pos (== 0, prefill path)

  const size_t per = (size_t)B_ * NH_ * S_ * HD_;   // 16M elems per tensor
  bf16* qws  = (bf16*)d_ws;
  bf16* kws  = qws + per;
  bf16* vtws = kws + per;    // V stored transposed: [B,NH,HD,S]
  bf16* ows  = vtws + per;

  dim3 blk(256);
  qkv_gemm_kernel<<<dim3(N3_ / 128, (B_ * S_) / 128), blk, 0, stream>>>(
      x, Wqkv, bqkv, qws, kws, vtws);
  attn_kernel<<<dim3(S_ / 128, B_ * NH_), blk, 0, stream>>>(qws, kws, vtws, ows);
  proj_gemm_kernel<<<dim3(DIM_ / 128, (B_ * S_) / 128), blk, 0, stream>>>(
      ows, Wproj, bproj, (float*)d_out);
}
